// Decoder_38740605010487
// MI455X (gfx1250) — matrix operand order repaired and checked by reading, not compiled, not run on hardware
//
#include <hip/hip_runtime.h>
#include <hip/hip_bf16.h>

// ---------------- problem constants ----------------
#define NLAYERS 3
#define BATCH   8192
#define H_DIM   1024
#define K_DIM   1024
#define DOUT    80
#define NACT    4096

// ---------------- GEMM tiling ----------------
#define BM 128          // rows of gate-space per block
#define BN 32           // gate columns per block
#define KC 32           // K per WMMA step
#define KCP 40          // padded LDS row stride (elements) -> 80B, conflict-free
#define THREADS 256     // 8 waves (wave32)

// dynamic-LDS layout (bytes), double buffered
#define SX_BYTES  (BM * KCP * 2)          // 10240
#define SH_OFF    SX_BYTES                // 10240
#define SW_OFF    (2 * SX_BYTES)          // 20480
#define SW_BYTES  (6 * BN * KCP * 2)      // 15360
#define BUF_BYTES (SW_OFF + SW_BYTES)     // 35840
#define SMEM_BYTES (2 * BUF_BYTES)        // 71680

typedef __attribute__((ext_vector_type(16))) __bf16 v16bf;
typedef __attribute__((ext_vector_type(8)))  float  v8f;

union FragBF { uint4 q[2]; v16bf v; };
union AccF   { v8f v; float f[8]; };

__device__ __forceinline__ float sigmoidf_(float x) {
    return __builtin_amdgcn_rcpf(1.0f + __expf(-x));
}

// CDNA5 async global->LDS copy (16B per lane), tracked by ASYNCcnt
__device__ __forceinline__ void async_b128(unsigned lds_byte_off, const void* gptr) {
    asm volatile("global_load_async_to_lds_b128 %0, %1, off"
                 :
                 : "v"(lds_byte_off), "v"(gptr)
                 : "memory");
}
__device__ __forceinline__ void wait_async0() {
    asm volatile("s_wait_asynccnt 0x0" ::: "memory");
}

// ---------------- helper kernels ----------------
__global__ __launch_bounds__(256)
void k_zero_f32(float* __restrict__ p, long n4) {      // n4 = count of float4
    long i = (long)blockIdx.x * blockDim.x + threadIdx.x;
    long stride = (long)gridDim.x * blockDim.x;
    float4 z = make_float4(0.f, 0.f, 0.f, 0.f);
    for (; i < n4; i += stride) ((float4*)p)[i] = z;
}

__global__ __launch_bounds__(256)
void k_f32_to_bf16(const float* __restrict__ s, __hip_bfloat16* __restrict__ d, long n4) {
    long i = (long)blockIdx.x * blockDim.x + threadIdx.x;
    long stride = (long)gridDim.x * blockDim.x;
    for (; i < n4; i += stride) {
        float4 v = ((const float4*)s)[i];
        union { __hip_bfloat16 h[4]; uint2 u; } pk;
        pk.h[0] = __float2bfloat16(v.x); pk.h[1] = __float2bfloat16(v.y);
        pk.h[2] = __float2bfloat16(v.z); pk.h[3] = __float2bfloat16(v.w);
        ((uint2*)d)[i] = pk.u;
    }
}

// gather rows: dst[m, :] = bf16(src[idx[m], :]) ; cols = K_DIM
__global__ __launch_bounds__(256)
void k_gather_bf16(const float* __restrict__ src, const int* __restrict__ idx,
                   __hip_bfloat16* __restrict__ dst, long n4) {
    long i = (long)blockIdx.x * blockDim.x + threadIdx.x;
    long stride = (long)gridDim.x * blockDim.x;
    for (; i < n4; i += stride) {
        long e = i * 4;
        int  m = (int)(e >> 10);          // / K_DIM
        int  c = (int)(e & (K_DIM - 1));
        float4 v = *(const float4*)(src + (size_t)idx[m] * K_DIM + c);
        union { __hip_bfloat16 h[4]; uint2 u; } pk;
        pk.h[0] = __float2bfloat16(v.x); pk.h[1] = __float2bfloat16(v.y);
        pk.h[2] = __float2bfloat16(v.z); pk.h[3] = __float2bfloat16(v.w);
        ((uint2*)dst)[i] = pk.u;
    }
}

// ---------------- fused GRU layer (bf16 WMMA, async double-buffered LDS) ----------------
__global__ __launch_bounds__(THREADS)
void k_gru_layer(const __hip_bfloat16* __restrict__ xb,   // [NACT, K]
                 const __hip_bfloat16* __restrict__ hb,   // [NACT, K]
                 const __hip_bfloat16* __restrict__ wih,  // [3H, K] bf16
                 const __hip_bfloat16* __restrict__ whh,  // [3H, K] bf16
                 const float* __restrict__ bih,           // [3H]
                 const float* __restrict__ bhh,           // [3H]
                 const float* __restrict__ hidden_l,      // [BATCH, H] f32 (original)
                 const int*   __restrict__ idx,           // [NACT]
                 float* __restrict__ newhid,              // [BATCH, H] f32 out (pre-copied)
                 __hip_bfloat16* __restrict__ xnext)      // [NACT, H] bf16 out
{
    extern __shared__ __align__(16) unsigned char smem[];

    const int tid  = threadIdx.x;
    const int m0   = blockIdx.x * BM;
    const int c0   = blockIdx.y * BN;
    const int wave = tid >> 5;
    const int lane = tid & 31;
    const int wm   = wave & 3;    // 32-row slab
    const int wn   = wave >> 2;   // 16-col slab
    const int lo   = lane & 15;
    const int hi   = lane >> 4;

    AccF acc[6][2];
#pragma unroll
    for (int g = 0; g < 6; ++g)
#pragma unroll
        for (int t = 0; t < 2; ++t)
#pragma unroll
            for (int v = 0; v < 8; ++v) acc[g][t].f[v] = 0.0f;

    // issue async stage of K-step `kt` into buffer at byte offset `base`
    auto stage = [&](unsigned base, int kt) {
#pragma unroll
        for (int j = 0; j < 2; ++j) {           // sx / sh: 512 16B chunks each
            int c = tid + THREADS * j;
            int r = c >> 2, kq = c & 3;
            unsigned doff = (unsigned)(r * 5 + kq) * 16u;
            async_b128(base + doff,          xb + (size_t)(m0 + r) * K_DIM + kt + kq * 8);
            async_b128(base + SH_OFF + doff, hb + (size_t)(m0 + r) * K_DIM + kt + kq * 8);
        }
#pragma unroll
        for (int j = 0; j < 3; ++j) {           // 6 weight tiles: 768 chunks
            int c = tid + THREADS * j;
            int t = c >> 7;                     // gate tile 0..5
            int r = (c >> 2) & 31;
            int kq = c & 3;
            const __hip_bfloat16* wsrc = (t < 3)
                ? (wih + (size_t)(c0 + t * H_DIM + r) * K_DIM + kt + kq * 8)
                : (whh + (size_t)(c0 + (t - 3) * H_DIM + r) * K_DIM + kt + kq * 8);
            async_b128(base + SW_OFF + (unsigned)((t * BN + r) * 5 + kq) * 16u, wsrc);
        }
    };

    const int NK = K_DIM / KC;   // 32
    stage(0u, 0);
    wait_async0();
    __syncthreads();

    for (int k = 0; k < NK; ++k) {
        const unsigned cur = (k & 1) ? (unsigned)BUF_BYTES : 0u;
        const unsigned nxt = (k & 1) ? 0u : (unsigned)BUF_BYTES;

        // prefetch next K-step into the other buffer (safe: prior barrier
        // guarantees every wave finished reading it)
        if (k + 1 < NK) stage(nxt, (k + 1) * KC);

        // ---- B fragments: lane holds W[c0+wn*16+lo, K 8*hi..+7 and 16+8*hi..+7] ----
        FragBF bf[6];
#pragma unroll
        for (int g = 0; g < 6; ++g) {
            const unsigned char* p = smem + cur + SW_OFF +
                (unsigned)((g * BN + wn * 16 + lo) * KCP + 8 * hi) * 2u;
            bf[g].q[0] = *(const uint4*)p;
            bf[g].q[1] = *(const uint4*)(p + 32);
        }

        // ---- A fragments + 12 WMMAs ----
#pragma unroll
        for (int t = 0; t < 2; ++t) {
            const int rloc = wm * 32 + t * 16 + lo;
            const unsigned aoff = (unsigned)(rloc * KCP + 8 * hi) * 2u;
            FragBF ax, ah;
            const unsigned char* px = smem + cur + aoff;
            ax.q[0] = *(const uint4*)px;
            ax.q[1] = *(const uint4*)(px + 32);
            const unsigned char* ph = smem + cur + SH_OFF + aoff;
            ah.q[0] = *(const uint4*)ph;
            ah.q[1] = *(const uint4*)(ph + 32);
#pragma unroll
            for (int g = 0; g < 3; ++g)
                acc[g][t].v = __builtin_amdgcn_wmma_f32_16x16x32_bf16(
                    false, ax.v, false, bf[g].v, (short)0, acc[g][t].v, false, false);
#pragma unroll
            for (int g = 3; g < 6; ++g)
                acc[g][t].v = __builtin_amdgcn_wmma_f32_16x16x32_bf16(
                    false, ah.v, false, bf[g].v, (short)0, acc[g][t].v, false, false);
        }

        wait_async0();       // our async copies for k+1 landed
        __syncthreads();     // everyone done computing cur + staging nxt
    }

    // ---- GRU gate epilogue (f32) ----
    const int c = c0 + wn * 16 + lo;             // hidden column, fixed per lane
    const float bir = bih[c],             bhr = bhh[c];
    const float biz = bih[H_DIM + c],     bhz = bhh[H_DIM + c];
    const float bin = bih[2 * H_DIM + c], bhn = bhh[2 * H_DIM + c];
#pragma unroll
    for (int t = 0; t < 2; ++t) {
#pragma unroll
        for (int v = 0; v < 8; ++v) {
            const int row  = m0 + wm * 32 + t * 16 + v + 8 * hi;
            const int gi   = idx[row];
            const float hp = hidden_l[(size_t)gi * H_DIM + c];
            const float r  = sigmoidf_(acc[0][t].f[v] + acc[3][t].f[v] + bir + bhr);
            const float z  = sigmoidf_(acc[1][t].f[v] + acc[4][t].f[v] + biz + bhz);
            const float n  = tanhf(acc[2][t].f[v] + bin + r * (acc[5][t].f[v] + bhn));
            const float ho = (1.0f - z) * n + z * hp;
            newhid[(size_t)gi * H_DIM + c] = ho;
            xnext[(size_t)row * H_DIM + c] = __float2bfloat16(ho);
        }
    }
}

// ---------------- head: y = tanh(x @ Wout^T + b), scattered ----------------
__global__ __launch_bounds__(THREADS)
void k_head(const __hip_bfloat16* __restrict__ xb,    // [NACT, 1024]
            const __hip_bfloat16* __restrict__ wout,  // [80, 1024] bf16
            const float* __restrict__ bout,           // [80]
            const int*   __restrict__ idx,
            float* __restrict__ out)                  // [BATCH, 80] (pre-zeroed)
{
    const int tid  = threadIdx.x;
    const int wave = tid >> 5;
    const int lane = tid & 31;
    const int lo   = lane & 15;
    const int hi   = lane >> 4;
    const int rbase = blockIdx.x * 128 + wave * 16;

    AccF acc[5];
#pragma unroll
    for (int nt = 0; nt < 5; ++nt)
#pragma unroll
        for (int v = 0; v < 8; ++v) acc[nt].f[v] = 0.0f;

    for (int kt = 0; kt < K_DIM; kt += KC) {
        FragBF a;
        const __hip_bfloat16* px = xb + (size_t)(rbase + lo) * K_DIM + kt + 8 * hi;
        a.q[0] = *(const uint4*)px;
        a.q[1] = *(const uint4*)(px + 16);
#pragma unroll
        for (int nt = 0; nt < 5; ++nt) {
            FragBF b;
            const __hip_bfloat16* pw = wout + (size_t)(nt * 16 + lo) * K_DIM + kt + 8 * hi;
            b.q[0] = *(const uint4*)pw;
            b.q[1] = *(const uint4*)(pw + 16);
            acc[nt].v = __builtin_amdgcn_wmma_f32_16x16x32_bf16(
                false, a.v, false, b.v, (short)0, acc[nt].v, false, false);
        }
    }
#pragma unroll
    for (int nt = 0; nt < 5; ++nt) {
        const int col = nt * 16 + lo;
        const float b = bout[col];
#pragma unroll
        for (int v = 0; v < 8; ++v) {
            const int row = rbase + v + 8 * hi;
            out[(size_t)idx[row] * DOUT + col] = tanhf(acc[nt].f[v] + b);
        }
    }
}

// ---------------- host launcher ----------------
extern "C" void kernel_launch(void* const* d_in, const int* in_sizes, int n_in,
                              void* d_out, int out_size, void* d_ws, size_t ws_size,
                              hipStream_t stream) {
    const float* encoded    = (const float*)d_in[0];
    const float* hidden     = (const float*)d_in[1];
    const int*   hidden_idx = (const int*)  d_in[2];
    const float* w_ih       = (const float*)d_in[3];
    const float* w_hh       = (const float*)d_in[4];
    const float* b_ih       = (const float*)d_in[5];
    const float* b_hh       = (const float*)d_in[6];
    const float* w_out      = (const float*)d_in[7];
    const float* b_out      = (const float*)d_in[8];

    float* out    = (float*)d_out;                       // [BATCH, DOUT]
    float* newhid = out + (size_t)BATCH * DOUT;          // [L, BATCH, H]

    // workspace carve-out (bf16 staging)
    char* ws = (char*)d_ws;
    size_t o = 0;
    auto carve = [&](size_t bytes) { void* p = ws + o; o += (bytes + 255) & ~(size_t)255; return p; };
    __hip_bfloat16* wihb  = (__hip_bfloat16*)carve((size_t)NLAYERS * 3 * H_DIM * K_DIM * 2);
    __hip_bfloat16* whhb  = (__hip_bfloat16*)carve((size_t)NLAYERS * 3 * H_DIM * K_DIM * 2);
    __hip_bfloat16* woutb = (__hip_bfloat16*)carve((size_t)DOUT * K_DIM * 2);
    __hip_bfloat16* xb0   = (__hip_bfloat16*)carve((size_t)NACT * K_DIM * 2);
    __hip_bfloat16* xb1   = (__hip_bfloat16*)carve((size_t)NACT * K_DIM * 2);
    __hip_bfloat16* hb    = (__hip_bfloat16*)carve((size_t)NACT * K_DIM * 2);

    // 1) zero the scattered output region
    k_zero_f32<<<640, 256, 0, stream>>>(out, (long)BATCH * DOUT / 4);

    // 2) bulk copy hidden -> new_hidden (rows get overwritten by GRU scatters)
    hipMemcpyAsync(newhid, hidden, (size_t)NLAYERS * BATCH * H_DIM * sizeof(float),
                   hipMemcpyDeviceToDevice, stream);

    // 3) weight conversion f32 -> bf16
    const long nw = (long)NLAYERS * 3 * H_DIM * K_DIM / 4;
    k_f32_to_bf16<<<2048, 256, 0, stream>>>(w_ih,  wihb,  nw);
    k_f32_to_bf16<<<2048, 256, 0, stream>>>(w_hh,  whhb,  nw);
    k_f32_to_bf16<<<64,   256, 0, stream>>>(w_out, woutb, (long)DOUT * K_DIM / 4);

    // 4) gather encoded[idx] -> x (bf16)
    k_gather_bf16<<<1024, 256, 0, stream>>>(encoded, hidden_idx, xb0, (long)NACT * K_DIM / 4);

    // 5) stacked GRU layers
    __hip_bfloat16* xcur = xb0;
    __hip_bfloat16* xnxt = xb1;
    for (int i = 0; i < NLAYERS; ++i) {
        const float* hidden_l = hidden + (size_t)i * BATCH * H_DIM;
        k_gather_bf16<<<1024, 256, 0, stream>>>(hidden_l, hidden_idx, hb, (long)NACT * K_DIM / 4);
        dim3 grid(NACT / BM, H_DIM / BN);
        k_gru_layer<<<grid, THREADS, SMEM_BYTES, stream>>>(
            xcur, hb,
            wihb + (size_t)i * 3 * H_DIM * K_DIM,
            whhb + (size_t)i * 3 * H_DIM * K_DIM,
            b_ih + (size_t)i * 3 * H_DIM,
            b_hh + (size_t)i * 3 * H_DIM,
            hidden_l, hidden_idx,
            newhid + (size_t)i * BATCH * H_DIM,
            xnxt);
        __hip_bfloat16* t = xcur; xcur = xnxt; xnxt = t;
    }

    // 6) output head
    k_head<<<NACT / 128, THREADS, 0, stream>>>(xcur, woutb, b_out, hidden_idx, out);
}
